// GHAT_89919435309272
// MI455X (gfx1250) — matrix operand order repaired and checked by reading, not compiled, not run on hardware
//
#include <hip/hip_runtime.h>

typedef __bf16 bf16;
typedef __attribute__((ext_vector_type(16))) __bf16 v16bf;
typedef __attribute__((ext_vector_type(8)))  __bf16 v8bf;
typedef __attribute__((ext_vector_type(8)))  float  v8f;

static constexpr int cB = 32, cN = 1024, cF = 128, cH = 8, cE = 128,
                     cFF = 512, cL = 4, cOUT = 64;

__device__ __forceinline__ bf16 f2bf(float x) { return (bf16)x; }

// ---------- block reduction over 128 threads (4 waves) ----------
__device__ __forceinline__ float block_reduce_128(float v, float* sbuf) {
#pragma unroll
  for (int o = 16; o > 0; o >>= 1) v += __shfl_down(v, o);
  if ((threadIdx.x & 31) == 0) sbuf[threadIdx.x >> 5] = v;
  __syncthreads();
  float r = sbuf[0] + sbuf[1] + sbuf[2] + sbuf[3];
  __syncthreads();
  return r;
}

// ---------- WMMA fragment loads (bf16, 16x16x32) ----------
// A 16x32 (MxK): lane holds row L%16; K = 8*(L/16)+0..7 and 16+8*(L/16)+0..7
__device__ __forceinline__ v16bf load_a_frag(const bf16* __restrict__ A, int lda,
                                             int row, int k) {
  const bf16* p = A + (size_t)row * lda + k;
  v8bf lo = *(const v8bf*)(p);
  v8bf hi = *(const v8bf*)(p + 16);
  return __builtin_shufflevector(lo, hi, 0, 1, 2, 3, 4, 5, 6, 7, 8, 9, 10, 11,
                                 12, 13, 14, 15);
}
// B 32x16 (KxN): lane holds col L%16; K = 8*(L/16)+0..7 and 16+8*(L/16)+0..7 (source row [N,K])
__device__ __forceinline__ v16bf load_b_frag(const bf16* __restrict__ B, int ldb,
                                             int row, int k) {
  const bf16* p = B + (size_t)row * ldb + k;
  v8bf lo = *(const v8bf*)(p);
  v8bf hi = *(const v8bf*)(p + 16);
  return __builtin_shufflevector(lo, hi, 0, 1, 2, 3, 4, 5, 6, 7, 8, 9, 10, 11,
                                 12, 13, 14, 15);
}

// ---------- GEMM: C = A[M,K] x Brow[N,K]^T, 128 thr / block, blocktile 128x128,
// wavetile 64x64. EPI: 0 = plain f32 store (batched), 1 = +bias, relu, bf16 store,
// 2 = +bias, f32 store.
template <int EPI>
__global__ __launch_bounds__(128) void gemm_bf16(
    const bf16* __restrict__ A, const bf16* __restrict__ B,
    void* __restrict__ Cout, const float* __restrict__ bias, int K, int lda,
    int ldb, int ldc, size_t bsA, size_t bsB, size_t bsC) {
  int wave = threadIdx.x >> 5;
  int lane = threadIdx.x & 31;
  int l16 = lane & 15, half = lane >> 4;
  int wm = wave >> 1, wn = wave & 1;
  int bm = blockIdx.y * 128 + wm * 64;
  int bn = blockIdx.x * 128 + wn * 64;
  A += (size_t)blockIdx.z * bsA;
  B += (size_t)blockIdx.z * bsB;

  v8f acc[4][4] = {};
  for (int k = 0; k < K; k += 32) {
    v16bf af[4], bfr[4];
#pragma unroll
    for (int i = 0; i < 4; i++)
      af[i] = load_a_frag(A, lda, bm + 16 * i + l16, k + 8 * half);
#pragma unroll
    for (int j = 0; j < 4; j++)
      bfr[j] = load_b_frag(B, ldb, bn + 16 * j + l16, k + 8 * half);
#pragma unroll
    for (int i = 0; i < 4; i++)
#pragma unroll
      for (int j = 0; j < 4; j++)
        acc[i][j] = __builtin_amdgcn_wmma_f32_16x16x32_bf16(
            false, af[i], false, bfr[j], (short)0, acc[i][j], false, false);
  }
#pragma unroll
  for (int i = 0; i < 4; i++) {
#pragma unroll
    for (int j = 0; j < 4; j++) {
      int col = bn + 16 * j + l16;
#pragma unroll
      for (int v = 0; v < 8; ++v) {
        int row = bm + 16 * i + v + 8 * half;
        float val = acc[i][j][v];
        if constexpr (EPI == 0) {
          ((float*)Cout)[(size_t)blockIdx.z * bsC + (size_t)row * ldc + col] = val;
        } else if constexpr (EPI == 1) {
          val += bias[col];
          val = val > 0.f ? val : 0.f;
          ((bf16*)Cout)[(size_t)row * ldc + col] = f2bf(val);
        } else {
          val += bias[col];
          ((float*)Cout)[(size_t)row * ldc + col] = val;
        }
      }
    }
  }
}

// ---------- one-time prep kernels ----------
// maskT[n,j] = (adj[j,n] > 0)
__global__ void mask_kernel(const int* __restrict__ adj, bf16* __restrict__ maskT) {
  int j = blockIdx.x;
  for (int n = threadIdx.x; n < cN; n += blockDim.x)
    maskT[(size_t)n * cN + j] = f2bf(adj[(size_t)j * cN + n] > 0 ? 1.f : 0.f);
}

__global__ void deg_kernel(const int* __restrict__ adj, float* __restrict__ deg) {
  int n = blockIdx.x * blockDim.x + threadIdx.x;
  if (n < cN) {
    int d = 0;
    for (int j = 0; j < cN; ++j) d += (adj[(size_t)j * cN + n] > 0);
    deg[n] = (float)d;
  }
}

__global__ void cvt_bf16_kernel(const float* __restrict__ in, bf16* __restrict__ out,
                                int count) {
  int i = blockIdx.x * blockDim.x + threadIdx.x;
  if (i < count) out[i] = f2bf(in[i]);
}

// copy input x -> xcur (f32) and XbfT (bf16 transposed [B,F,N])
__global__ void xprep_kernel(const float* __restrict__ xin, float* __restrict__ xcur,
                             bf16* __restrict__ xbfT) {
  int bn = blockIdx.x, b = bn >> 10, n = bn & 1023, f = threadIdx.x;
  size_t ix = (size_t)bn * cF + f;
  float v = xin[ix];
  xcur[ix] = v;
  xbfT[((size_t)b * cF + f) * cN + n] = f2bf(v);
}

// ---------- per-layer small kernels ----------
// C{1,2}[h,n,f] = sum_e Ws[h,e,f] * As[h, sel*E+e, n]
__global__ void cmat_kernel(const float* __restrict__ Ws_l,
                            const float* __restrict__ As_l, float* __restrict__ C1,
                            float* __restrict__ C2) {
  int n = blockIdx.x, h = blockIdx.y, sel = blockIdx.z, f = threadIdx.x;
  const float* W = Ws_l + (size_t)h * cE * cF;
  const float* A = As_l + ((size_t)h * 2 * cE + sel * cE) * cN + n;
  float acc = 0.f;
  for (int e = 0; e < cE; ++e) acc += W[(size_t)e * cF + f] * A[(size_t)e * cN];
  float* C = sel ? C2 : C1;
  C[((size_t)h * cN + n) * cF + f] = acc;
}

// sb[h,n] = bs[h]:As1[h,:,n] + deg[n] * bs[h]:As2[h,:,n]
__global__ void sbias_kernel(const float* __restrict__ bs_l,
                             const float* __restrict__ As_l,
                             const float* __restrict__ deg, float* __restrict__ sb) {
  int h = blockIdx.y;
  int n = blockIdx.x * 128 + threadIdx.x;
  float a1 = 0.f, a2 = 0.f;
  for (int e = 0; e < cE; ++e) {
    float bv = bs_l[h * cE + e];
    a1 += bv * As_l[((size_t)h * 2 * cE + e) * cN + n];
    a2 += bv * As_l[((size_t)h * 2 * cE + cE + e) * cN + n];
  }
  sb[h * cN + n] = a1 + deg[n] * a2;
}

// xs[b,f] = sum_n x[b,n,f]
__global__ void xs_kernel(const float* __restrict__ x, float* __restrict__ xs) {
  int b = blockIdx.x, f = threadIdx.x;
  float a = 0.f;
  for (int n = 0; n < cN; ++n) a += x[((size_t)b * cN + n) * cF + f];
  xs[b * cF + f] = a;
}

// hsum[b,h,e] = sum_f Ws[h,e,f]*xs[b,f] + N*bs[h,e]
__global__ void hsum_kernel(const float* __restrict__ Ws_l,
                            const float* __restrict__ bs_l,
                            const float* __restrict__ xs, float* __restrict__ hsum) {
  int b = blockIdx.x, h = blockIdx.y, e = threadIdx.x;
  __shared__ float xsv[cF];
  xsv[e] = xs[b * cF + e];
  __syncthreads();
  const float* W = Ws_l + ((size_t)h * cE + e) * cF;
  float a = 0.f;
  for (int f = 0; f < cF; ++f) a += W[f] * xsv[f];
  hsum[((size_t)b * cH + h) * cF + e] = a + (float)cN * bs_l[h * cE + e];
}

// fused: score (8 block-dots), attention + leaky_relu + head-sum, residual, LN1
__global__ void score_att_ln1_kernel(
    const float* __restrict__ x, const float* __restrict__ Y,
    const float* __restrict__ C1, const float* __restrict__ C2,
    const float* __restrict__ sb, const float* __restrict__ hsum,
    const float* __restrict__ g1, const float* __restrict__ be1,
    float* __restrict__ x1, bf16* __restrict__ x1bf) {
  int bn = blockIdx.x, b = bn >> 10, n = bn & 1023, f = threadIdx.x;
  __shared__ float sred[4];
  __shared__ float sc[cH];
  size_t ix = (size_t)bn * cF + f;
  float xv = x[ix], yv = Y[ix];
  for (int h = 0; h < cH; ++h) {
    float p = xv * C1[((size_t)h * cN + n) * cF + f] +
              yv * C2[((size_t)h * cN + n) * cF + f];
    float s = block_reduce_128(p, sred);
    if (threadIdx.x == 0) sc[h] = s + sb[h * cN + n];
  }
  __syncthreads();
  float att = 0.f;
#pragma unroll
  for (int h = 0; h < cH; ++h) {
    float v = sc[h] * hsum[((size_t)b * cH + h) * cF + f];
    att += v > 0.f ? v : 0.01f * v;
  }
  float r = xv + att;
  float m = block_reduce_128(r, sred) * (1.f / cF);
  float d = r - m;
  float var = block_reduce_128(d * d, sred) * (1.f / cF);
  float xo = d * rsqrtf(var + 1e-5f) * g1[f] + be1[f];
  x1[ix] = xo;
  x1bf[ix] = f2bf(xo);
}

// x = LN(x1 + U); emit f32 and transposed bf16 for next layer's Y GEMM
__global__ void ln2_kernel(const float* __restrict__ x1, const float* __restrict__ U,
                           const float* __restrict__ g2,
                           const float* __restrict__ be2, float* __restrict__ xout,
                           bf16* __restrict__ xbfT) {
  int bn = blockIdx.x, b = bn >> 10, n = bn & 1023, f = threadIdx.x;
  __shared__ float sred[4];
  size_t ix = (size_t)bn * cF + f;
  float r = x1[ix] + U[ix];
  float m = block_reduce_128(r, sred) * (1.f / cF);
  float d = r - m;
  float var = block_reduce_128(d * d, sred) * (1.f / cF);
  float xo = d * rsqrtf(var + 1e-5f) * g2[f] + be2[f];
  xout[ix] = xo;
  xbfT[((size_t)b * cF + f) * cN + n] = f2bf(xo);
}

// out[b,o] = bo[o] + sum_k relu(x[b,k]) * Wo[o,k]
__global__ void out_kernel(const float* __restrict__ xf, const float* __restrict__ Wo,
                           const float* __restrict__ bo, float* __restrict__ out) {
  int o = blockIdx.x, b = blockIdx.y;
  const int Kt = cN * cF;
  const float* xr = xf + (size_t)b * Kt;
  const float* wr = Wo + (size_t)o * Kt;
  float a = 0.f;
  for (int k = threadIdx.x; k < Kt; k += 256) {
    float v = xr[k];
    v = v > 0.f ? v : 0.f;
    a += v * wr[k];
  }
#pragma unroll
  for (int off = 16; off > 0; off >>= 1) a += __shfl_down(a, off);
  __shared__ float s[8];
  if ((threadIdx.x & 31) == 0) s[threadIdx.x >> 5] = a;
  __syncthreads();
  if (threadIdx.x == 0) {
    float t = 0.f;
#pragma unroll
    for (int i = 0; i < 8; ++i) t += s[i];
    out[(size_t)b * cOUT + o] = t + bo[o];
  }
}

extern "C" void kernel_launch(void* const* d_in, const int* in_sizes, int n_in,
                              void* d_out, int out_size, void* d_ws, size_t ws_size,
                              hipStream_t stream) {
  const float* x_in = (const float*)d_in[0];
  const int* adj = (const int*)d_in[1];
  const float* Ws = (const float*)d_in[2];
  const float* bs = (const float*)d_in[3];
  const float* As = (const float*)d_in[4];
  const float* W1 = (const float*)d_in[5];
  const float* b1 = (const float*)d_in[6];
  const float* W2 = (const float*)d_in[7];
  const float* b2 = (const float*)d_in[8];
  const float* g1 = (const float*)d_in[9];
  const float* be1 = (const float*)d_in[10];
  const float* g2 = (const float*)d_in[11];
  const float* be2 = (const float*)d_in[12];
  const float* Wo = (const float*)d_in[13];
  const float* bo = (const float*)d_in[14];

  char* w = (char*)d_ws;
  auto alloc = [&](size_t bytes) {
    char* p = w;
    w += (bytes + 255) & ~(size_t)255;
    return p;
  };
  bf16* maskT = (bf16*)alloc((size_t)cN * cN * 2);
  bf16* XbfT = (bf16*)alloc((size_t)cB * cF * cN * 2);
  float* xcur = (float*)alloc((size_t)cB * cN * cF * 4);
  float* Y = (float*)alloc((size_t)cB * cN * cF * 4);
  float* C1 = (float*)alloc((size_t)cH * cN * cF * 4);
  float* C2 = (float*)alloc((size_t)cH * cN * cF * 4);
  float* sb = (float*)alloc((size_t)cH * cN * 4);
  float* deg = (float*)alloc((size_t)cN * 4);
  float* xs = (float*)alloc((size_t)cB * cF * 4);
  float* hsum = (float*)alloc((size_t)cB * cH * cF * 4);
  float* x1 = (float*)alloc((size_t)cB * cN * cF * 4);
  bf16* x1bf = (bf16*)alloc((size_t)cB * cN * cF * 2);
  bf16* T = (bf16*)alloc((size_t)cB * cN * cFF * 2);
  float* U = (float*)alloc((size_t)cB * cN * cF * 4);
  bf16* W1bf = (bf16*)alloc((size_t)cL * cFF * cF * 2);
  bf16* W2bf = (bf16*)alloc((size_t)cL * cF * cFF * 2);

  // one-time prep
  mask_kernel<<<cN, 256, 0, stream>>>(adj, maskT);
  deg_kernel<<<cN / 256, 256, 0, stream>>>(adj, deg);
  {
    int cnt = cL * cFF * cF;
    cvt_bf16_kernel<<<(cnt + 255) / 256, 256, 0, stream>>>(W1, W1bf, cnt);
    cvt_bf16_kernel<<<(cnt + 255) / 256, 256, 0, stream>>>(W2, W2bf, cnt);
  }
  xprep_kernel<<<cB * cN, cF, 0, stream>>>(x_in, xcur, XbfT);

  for (int l = 0; l < cL; ++l) {
    const float* Ws_l = Ws + (size_t)l * cH * cE * cF;
    const float* bs_l = bs + (size_t)l * cH * cE;
    const float* As_l = As + (size_t)l * cH * 2 * cE * cN;
    const bf16* W1bf_l = W1bf + (size_t)l * cFF * cF;
    const bf16* W2bf_l = W2bf + (size_t)l * cF * cFF;
    const float* b1_l = b1 + (size_t)l * cFF;
    const float* b2_l = b2 + (size_t)l * cF;

    cmat_kernel<<<dim3(cN, cH, 2), cF, 0, stream>>>(Ws_l, As_l, C1, C2);
    sbias_kernel<<<dim3(cN / 128, cH), 128, 0, stream>>>(bs_l, As_l, deg, sb);
    xs_kernel<<<cB, cF, 0, stream>>>(xcur, xs);
    hsum_kernel<<<dim3(cB, cH), cF, 0, stream>>>(Ws_l, bs_l, xs, hsum);

    // Y[b] = maskT (1024x1024) @ x[b] (1024x128), via B-operand = XbfT[b] [F,N]
    gemm_bf16<0><<<dim3(1, cN / 128, cB), 128, 0, stream>>>(
        maskT, XbfT, Y, nullptr, /*K=*/cN, /*lda=*/cN, /*ldb=*/cN, /*ldc=*/cF,
        /*bsA=*/0, /*bsB=*/(size_t)cF * cN, /*bsC=*/(size_t)cN * cF);

    score_att_ln1_kernel<<<cB * cN, cF, 0, stream>>>(
        xcur, Y, C1, C2, sb, hsum, g1 + (size_t)l * cF, be1 + (size_t)l * cF, x1,
        x1bf);

    // T = relu(x1 @ W1^T + b1)  [32768 x 512], bf16 out
    gemm_bf16<1><<<dim3(cFF / 128, (cB * cN) / 128, 1), 128, 0, stream>>>(
        x1bf, W1bf_l, T, b1_l, /*K=*/cF, /*lda=*/cF, /*ldb=*/cF, /*ldc=*/cFF, 0, 0,
        0);
    // U = T @ W2^T + b2  [32768 x 128], f32 out
    gemm_bf16<2><<<dim3(cF / 128, (cB * cN) / 128, 1), 128, 0, stream>>>(
        T, W2bf_l, U, b2_l, /*K=*/cFF, /*lda=*/cFF, /*ldb=*/cFF, /*ldc=*/cF, 0, 0,
        0);

    ln2_kernel<<<cB * cN, cF, 0, stream>>>(x1, U, g2 + (size_t)l * cF,
                                           be2 + (size_t)l * cF, xcur, XbfT);
  }

  out_kernel<<<dim3(cOUT, cB), 256, 0, stream>>>(xcur, Wo, bo, (float*)d_out);
}
